// hypergraph_Undirected_88484916232596
// MI455X (gfx1250) — matrix operand order repaired and checked by reading, not compiled, not run on hardware
//
#include <hip/hip_runtime.h>
#include <hip/hip_bf16.h>
#include <math.h>

#define NNODES 8192
#define DIM 64
#define KTOP 64
#define ALPHA 3.0f
#define SIM_TH 0.5f
#define CHUNK 128            // columns per chunk (8 waves x 16 cols)

typedef __attribute__((ext_vector_type(16))) _Float16 v16h;
typedef __attribute__((ext_vector_type(8)))  float    v8f;

union Frag16 { v16h h; unsigned u[8]; };

// ---------------------------------------------------------------------------
// Kernel 1: h = tanh(alpha*(emb @ W^T + b)); hn = h / max(||h||, 1e-8)  -> f16
// One wave (32 lanes) per node; lane handles dims (l, l+32).
// ---------------------------------------------------------------------------
__global__ __launch_bounds__(256) void feat_norm_kernel(
    const float* __restrict__ emb, const float* __restrict__ lw,
    const float* __restrict__ lb, _Float16* __restrict__ hn) {
  __shared__ float lds_e[8][DIM];
  const int lane = threadIdx.x & 31;
  const int wv   = threadIdx.x >> 5;
  const int n    = blockIdx.x * 8 + wv;

  lds_e[wv][lane]      = emb[n * DIM + lane];
  lds_e[wv][lane + 32] = emb[n * DIM + lane + 32];
  __syncthreads();

  float acc0 = lb[lane];
  float acc1 = lb[lane + 32];
#pragma unroll 8
  for (int i = 0; i < DIM; ++i) {
    const float e = lds_e[wv][i];
    acc0 = fmaf(e, lw[lane * DIM + i], acc0);
    acc1 = fmaf(e, lw[(lane + 32) * DIM + i], acc1);
  }
  const float h0 = tanhf(ALPHA * acc0);
  const float h1 = tanhf(ALPHA * acc1);

  float sq = h0 * h0 + h1 * h1;                  // wave32 reduction
#pragma unroll
  for (int off = 16; off > 0; off >>= 1) sq += __shfl_xor(sq, off, 32);
  const float inv = 1.0f / fmaxf(sqrtf(sq), 1e-8f);

  hn[n * DIM + lane]      = (_Float16)(h0 * inv);
  hn[n * DIM + lane + 32] = (_Float16)(h1 * inv);
}

// ---------------------------------------------------------------------------
// Kernel 2: fused sim = hn @ hn^T (WMMA f16, f32 acc), threshold, running
// top-64 per row. Skip invariant: chunks processed in increasing column
// order, so after chunk 0 the pool holds 64 real entries, and any chunk with
// no value >= 0.5 cannot change the top-64 (later zeros lose the index
// tie-break). One workgroup per 16-row tile; 8 waves x 16-col WMMA tiles.
// ---------------------------------------------------------------------------
__global__ __launch_bounds__(256) void sim_topk_kernel(
    const _Float16* __restrict__ hn, float* __restrict__ H) {
  const int tid  = threadIdx.x;
  const int lane = tid & 31;
  const int wv   = tid >> 5;
  const int hi   = (lane & 16) ? 1 : 0;
  const int i0   = blockIdx.x * 16;
  const int r    = tid >> 4;       // row (0..15) this thread helps sort
  const int t    = tid & 15;       // worker id within row

  __shared__ unsigned long long pool[16][256];   // [0..63] running, [64..191] new, [192..255] pad
  __shared__ int flag;

  // init running top lists to "empty" (key 0 loses to every real candidate)
  for (int z = tid; z < 16 * 64; z += 256) pool[z >> 6][z & 63] = 0ULL;

  const unsigned* hn_u = (const unsigned*)hn;    // f16 pairs as dwords

  // --- A fragments (rows i0..i0+15, K=0..63), ISA 16-bit 16x32 layout:
  // lane<16: M=lane, VGPR v holds K pair base (v<4 ? 2v : 16+2(v-4)); hi lanes +8
  const int arow = i0 + (lane & 15);
  Frag16 a0, a1;
#pragma unroll
  for (int v = 0; v < 8; ++v) {
    const int ui = ((v < 4) ? v : v + 4) + (hi ? 4 : 0);  // dword index in row
    a0.u[v] = hn_u[arow * 32 + ui];        // K 0..31
    a1.u[v] = hn_u[arow * 32 + 16 + ui];   // K 32..63
  }

  for (int ch = 0; ch < NNODES / CHUNK; ++ch) {
    const int j0 = ch * CHUNK;

    // prefetch next chunk's B columns (L2-resident; warms WGP$)
    if (ch < NNODES / CHUNK - 1)
      __builtin_prefetch(hn + (j0 + CHUNK) * DIM + tid * 32, 0, 1);

    if (tid == 0) flag = 0;
    __syncthreads();

    // --- B fragments: 32x16, lane&15 = column, K striped over VGPRs
    const int jc = j0 + wv * 16 + (lane & 15);
    Frag16 b0, b1;
#pragma unroll
    for (int v = 0; v < 8; ++v) {
      b0.u[v] = hn_u[jc * 32 + ((v < 4) ? v : v + 4) + (hi ? 4 : 0)];        // K 0..31
      b1.u[v] = hn_u[jc * 32 + 16 + ((v < 4) ? v : v + 4) + (hi ? 4 : 0)];   // K 32..63
    }

    v8f c = {};
    c = __builtin_amdgcn_wmma_f32_16x16x32_f16(false, a0.h, false, b0.h,
                                               (short)0, c, false, false);
    c = __builtin_amdgcn_wmma_f32_16x16x32_f16(false, a1.h, false, b1.h,
                                               (short)0, c, false, false);

    // threshold + encode (value desc, index asc) into 64-bit keys in registers
    const int lc = wv * 16 + (lane & 15);
    const unsigned jcol = (unsigned)(j0 + lc);
    unsigned long long keys[8];
    bool any = false;
#pragma unroll
    for (int p = 0; p < 8; ++p) {
      float s = c[p];
      const bool keep = (s >= SIM_TH);
      any = any || keep;
      if (!keep) s = 0.0f;
      keys[p] = ((unsigned long long)__float_as_uint(s) << 32) |
                (0xFFFFFFFFu - jcol);
    }
    if (any) flag = 1;          // benign race: everyone stores 1
    __syncthreads();
    const bool doMerge = (ch == 0) || (flag != 0);
    __syncthreads();            // all reads of flag done before next reset
    if (!doMerge) continue;     // uniform branch

    // spill candidates + zero pad slots
#pragma unroll
    for (int p = 0; p < 8; ++p) pool[p + (hi ? 8 : 0)][64 + lc] = keys[p];
    for (int z = tid; z < 16 * 64; z += 256) pool[z >> 6][192 + (z & 63)] = 0ULL;
    __syncthreads();

    // descending bitonic sort of 256 keys per row (16 threads per row)
    for (unsigned kk = 2; kk <= 256; kk <<= 1) {
      for (unsigned jj = kk >> 1; jj > 0; jj >>= 1) {
        for (unsigned x = t; x < 256; x += 16) {
          const unsigned ixj = x ^ jj;
          if (ixj > x) {
            const unsigned long long A = pool[r][x];
            const unsigned long long B = pool[r][ixj];
            const bool up = ((x & kk) == 0);
            if (up ? (A < B) : (A > B)) { pool[r][x] = B; pool[r][ixj] = A; }
          }
        }
        __syncthreads();
      }
    }
  }
  __syncthreads();

  // scatter: slot k of row i0+r -> H[k, idx] = 1 (races store identical 1.0f)
  for (int k = t; k < KTOP; k += 16) {
    const unsigned long long key = pool[r][k];
    const unsigned idxj = 0xFFFFFFFFu - (unsigned)(key & 0xFFFFFFFFull);
    H[k * NNODES + idxj] = 1.0f;
  }
}

// ---------------------------------------------------------------------------
extern "C" void kernel_launch(void* const* d_in, const int* in_sizes, int n_in,
                              void* d_out, int out_size, void* d_ws, size_t ws_size,
                              hipStream_t stream) {
  (void)in_sizes; (void)n_in; (void)out_size; (void)ws_size;
  // inputs: idx(int64, identity arange — unused), emb_weight, lin_w, lin_b
  const float* emb = (const float*)d_in[1];
  const float* lw  = (const float*)d_in[2];
  const float* lb  = (const float*)d_in[3];
  _Float16*    hn  = (_Float16*)d_ws;            // [8192][64] f16 = 1 MB scratch
  float*       H   = (float*)d_out;              // [64][8192] f32

  hipMemsetAsync(d_out, 0, (size_t)KTOP * NNODES * sizeof(float), stream);
  feat_norm_kernel<<<NNODES / 8, 256, 0, stream>>>(emb, lw, lb, hn);
  sim_topk_kernel<<<NNODES / 16, 256, 0, stream>>>(hn, H);
}
